// TimeSeriesModel_1Step_ES_309237645776
// MI455X (gfx1250) — matrix operand order repaired and checked by reading, not compiled, not run on hardware
//
#include <hip/hip_runtime.h>
#include <hip/hip_bf16.h>
#include <math.h>

#define ALPHA 0.3f
#define B_ 256
#define T_ 1024
#define I_ 128
#define H_ 1024

typedef __attribute__((ext_vector_type(16))) __bf16 v16bf;
typedef __attribute__((ext_vector_type(8)))  float  v8f;
typedef __attribute__((ext_vector_type(4)))  unsigned int v4u;

union FragAB {
    v16bf v;
    v4u   q[2];
    unsigned short u[16];
};

__device__ __forceinline__ unsigned short f32_to_bf16(float f) {
    unsigned int u = __float_as_uint(f);
    unsigned int r = u + 0x7FFFu + ((u >> 16) & 1u);   // round-to-nearest-even
    return (unsigned short)(r >> 16);
}

__device__ __forceinline__ float fast_sigmoid(float x) {
    return 1.f / (1.f + __expf(-x));
}
__device__ __forceinline__ float fast_tanh(float x) {
    float e = __expf(2.f * x);
    return 1.f - 2.f / (e + 1.f);
}

#define WMMA_BF16(A, Bm, C) \
    __builtin_amdgcn_wmma_f32_16x16x32_bf16(false, (A), false, (Bm), (short)0, (C), false, false)

// ---------------------------------------------------------------------------
// Kernel 1: EMA smoothing scan over T; emit bf16 smoothed series [T,B,I] and
// the final smoothed step st1 [B,I] in f32.
// ---------------------------------------------------------------------------
__global__ void ema_kernel(const float* __restrict__ x,
                           unsigned short* __restrict__ sm,
                           float* __restrict__ st1) {
    int idx = blockIdx.x * blockDim.x + threadIdx.x;   // b*I + i
    int b = idx / I_;
    int i = idx % I_;
    const float* xp = x + (size_t)b * T_ * I_ + i;
    float s = xp[0];
    sm[idx] = f32_to_bf16(s);
    for (int t = 1; t < T_; ++t) {
        float xv = xp[(size_t)t * I_];
        s = (1.f - ALPHA) * xv + ALPHA * s;
        sm[(size_t)t * (B_ * I_) + idx] = f32_to_bf16(s);
    }
    st1[idx] = s;
}

// ---------------------------------------------------------------------------
// Kernel 2: init h0 = 0 (f32 + bf16 copies) and reset grid-barrier state.
// ---------------------------------------------------------------------------
__global__ void init_kernel(float* __restrict__ hF,
                            unsigned short* __restrict__ hB,
                            int* __restrict__ sync) {
    int idx = blockIdx.x * blockDim.x + threadIdx.x;
    if (idx < B_ * H_) {
        hF[idx] = 0.f;
        hB[idx] = 0;
    }
    if (idx < 8) sync[idx] = 0;
}

// ---------------------------------------------------------------------------
// Grid-wide barrier (persistent kernel, all blocks co-resident).
// ---------------------------------------------------------------------------
__device__ __forceinline__ void grid_barrier(int* cnt, int* gen, int nblk) {
    __threadfence();
    __syncthreads();
    if (threadIdx.x == 0) {
        int g = atomicAdd(gen, 0);
        int old = atomicAdd(cnt, 1);
        if (old == nblk - 1) {
            atomicExch(cnt, 0);
            __threadfence();
            atomicAdd(gen, 1);
        } else {
            while (atomicAdd(gen, 0) == g) { __builtin_amdgcn_s_sleep(1); }
        }
    }
    __syncthreads();
}

// ---------------------------------------------------------------------------
// Kernel 3: persistent fused GRU. 64 blocks; block c owns hidden columns
// [16c, 16c+16) and computes gates r/z/n for those columns every step.
// Single K sweep keeps BOTH row-tiles' accumulators live so each LDS
// B-fragment feeds two WMMAs (1 WMMA per b128 load).
// ---------------------------------------------------------------------------
#define GRU_BLOCKS 64
#define WHH_LD (H_ + 8)   // pad: row stride 2064B dodges LDS bank conflicts
#define WIH_LD (I_ + 8)

__launch_bounds__(256, 1)
__global__ void gru_kernel(const unsigned short* __restrict__ sm,
                           const float* __restrict__ Wih,
                           const float* __restrict__ Whh,
                           const float* __restrict__ bih,
                           const float* __restrict__ bhh,
                           float* __restrict__ hF,
                           unsigned short* __restrict__ hB,
                           int* __restrict__ sync) {
    __shared__ unsigned short ldsWhh[3][16][WHH_LD];   // ~97 KB
    __shared__ unsigned short ldsWih[3][16][WIH_LD];   // ~13 KB

    const int c    = blockIdx.x;        // hidden column strip
    const int col0 = c * 16;

    // ---- stage weight strips into LDS as bf16 (once) ----
    for (int e = threadIdx.x; e < 3 * 16 * H_; e += blockDim.x) {
        int gate = e / (16 * H_);
        int rem  = e % (16 * H_);
        int n    = rem / H_;
        int k    = rem % H_;
        ldsWhh[gate][n][k] =
            f32_to_bf16(Whh[(size_t)(gate * H_ + col0 + n) * H_ + k]);
    }
    for (int e = threadIdx.x; e < 3 * 16 * I_; e += blockDim.x) {
        int gate = e / (16 * I_);
        int rem  = e % (16 * I_);
        int n    = rem / I_;
        int k    = rem % I_;
        ldsWih[gate][n][k] =
            f32_to_bf16(Wih[(size_t)(gate * H_ + col0 + n) * I_ + k]);
    }
    __syncthreads();

    const int lane   = threadIdx.x & 31;
    const int wave   = threadIdx.x >> 5;
    const int cl     = lane & 15;        // C/D column within tile, B-matrix N
    const int hiHalf = lane >> 4;        // K-half selector for A/B fragments
    const int ch     = col0 + cl;        // global hidden column

    const float br  = bih[ch] + bhh[ch];
    const float bz  = bih[H_ + ch] + bhh[H_ + ch];
    const float bin = bih[2 * H_ + ch];
    const float bhn = bhh[2 * H_ + ch];

    const int m0a = wave * 16;           // row tile 0 of this wave
    const int m0b = (wave + 8) * 16;     // row tile 1 of this wave
    const int rowA = m0a + cl;           // A-matrix row this lane services
    const int rowB = m0b + cl;

    for (int t = 0; t < T_; ++t) {
        const int pr = t & 1;
        const unsigned short* hcurB = hB + (size_t)pr * B_ * H_;
        const float*          hcurF = hF + (size_t)pr * B_ * H_;
        float*          hnxtF = hF + (size_t)(1 - pr) * B_ * H_;
        unsigned short* hnxtB = hB + (size_t)(1 - pr) * B_ * H_;
        const unsigned short* smt = sm + (size_t)t * B_ * I_;

        // prefetch next step's smoothed-input slab (64 KB, fully covered)
        if (t + 1 < T_) {
            __builtin_prefetch(sm + (size_t)(t + 1) * B_ * I_ +
                               (size_t)threadIdx.x * 128, 0, 1);
        }

        v8f accR0  = {}, accZ0  = {}, accIN0 = {}, accHN0 = {};
        v8f accR1  = {}, accZ1  = {}, accIN1 = {}, accHN1 = {};

        // ---- input-GEMM part: A = sm_t (256 x 128), K = 128 ----
        {
            const unsigned short* aR0 = smt + (size_t)rowA * I_;
            const unsigned short* aR1 = smt + (size_t)rowB * I_;
            for (int k0 = 0; k0 < I_; k0 += 32) {
                FragAB bR, bZ, bN;
                bR.q[0] = *(const v4u*)(&ldsWih[0][cl][k0 + hiHalf * 16]);
                bR.q[1] = *(const v4u*)(&ldsWih[0][cl][k0 + hiHalf * 16 + 8]);
                bZ.q[0] = *(const v4u*)(&ldsWih[1][cl][k0 + hiHalf * 16]);
                bZ.q[1] = *(const v4u*)(&ldsWih[1][cl][k0 + hiHalf * 16 + 8]);
                bN.q[0] = *(const v4u*)(&ldsWih[2][cl][k0 + hiHalf * 16]);
                bN.q[1] = *(const v4u*)(&ldsWih[2][cl][k0 + hiHalf * 16 + 8]);
                FragAB a0, a1;
                a0.q[0] = *(const v4u*)(aR0 + k0 + hiHalf * 16);
                a0.q[1] = *(const v4u*)(aR0 + k0 + hiHalf * 16 + 8);
                a1.q[0] = *(const v4u*)(aR1 + k0 + hiHalf * 16);
                a1.q[1] = *(const v4u*)(aR1 + k0 + hiHalf * 16 + 8);
                accR0  = WMMA_BF16(a0.v, bR.v, accR0);
                accR1  = WMMA_BF16(a1.v, bR.v, accR1);
                accZ0  = WMMA_BF16(a0.v, bZ.v, accZ0);
                accZ1  = WMMA_BF16(a1.v, bZ.v, accZ1);
                accIN0 = WMMA_BF16(a0.v, bN.v, accIN0);
                accIN1 = WMMA_BF16(a1.v, bN.v, accIN1);
            }
        }

        // ---- recurrent-GEMM part: A = h_t (256 x 1024), K = 1024 ----
        {
            const unsigned short* aR0 = hcurB + (size_t)rowA * H_;
            const unsigned short* aR1 = hcurB + (size_t)rowB * H_;
            for (int k0 = 0; k0 < H_; k0 += 32) {
                FragAB bR, bZ, bN;
                bR.q[0] = *(const v4u*)(&ldsWhh[0][cl][k0 + hiHalf * 16]);
                bR.q[1] = *(const v4u*)(&ldsWhh[0][cl][k0 + hiHalf * 16 + 8]);
                bZ.q[0] = *(const v4u*)(&ldsWhh[1][cl][k0 + hiHalf * 16]);
                bZ.q[1] = *(const v4u*)(&ldsWhh[1][cl][k0 + hiHalf * 16 + 8]);
                bN.q[0] = *(const v4u*)(&ldsWhh[2][cl][k0 + hiHalf * 16]);
                bN.q[1] = *(const v4u*)(&ldsWhh[2][cl][k0 + hiHalf * 16 + 8]);
                FragAB a0, a1;
                a0.q[0] = *(const v4u*)(aR0 + k0 + hiHalf * 16);
                a0.q[1] = *(const v4u*)(aR0 + k0 + hiHalf * 16 + 8);
                a1.q[0] = *(const v4u*)(aR1 + k0 + hiHalf * 16);
                a1.q[1] = *(const v4u*)(aR1 + k0 + hiHalf * 16 + 8);
                accR0  = WMMA_BF16(a0.v, bR.v, accR0);
                accR1  = WMMA_BF16(a1.v, bR.v, accR1);
                accZ0  = WMMA_BF16(a0.v, bZ.v, accZ0);
                accZ1  = WMMA_BF16(a1.v, bZ.v, accZ1);
                accHN0 = WMMA_BF16(a0.v, bN.v, accHN0);
                accHN1 = WMMA_BF16(a1.v, bN.v, accHN1);
            }
        }

        // ---- gates + state update (C/D layout: VGPR j -> row m0+j+8*hi) ----
        #pragma unroll
        for (int j = 0; j < 8; ++j) {
            int m = m0a + j + hiHalf * 8;
            float r  = fast_sigmoid(accR0[j] + br);
            float z  = fast_sigmoid(accZ0[j] + bz);
            float nv = fast_tanh(accIN0[j] + bin + r * (accHN0[j] + bhn));
            float hp = hcurF[(size_t)m * H_ + ch];
            float hn = (1.f - z) * nv + z * hp;
            hnxtF[(size_t)m * H_ + ch] = hn;
            hnxtB[(size_t)m * H_ + ch] = f32_to_bf16(hn);
        }
        #pragma unroll
        for (int j = 0; j < 8; ++j) {
            int m = m0b + j + hiHalf * 8;
            float r  = fast_sigmoid(accR1[j] + br);
            float z  = fast_sigmoid(accZ1[j] + bz);
            float nv = fast_tanh(accIN1[j] + bin + r * (accHN1[j] + bhn));
            float hp = hcurF[(size_t)m * H_ + ch];
            float hn = (1.f - z) * nv + z * hp;
            hnxtF[(size_t)m * H_ + ch] = hn;
            hnxtB[(size_t)m * H_ + ch] = f32_to_bf16(hn);
        }

        grid_barrier(&sync[0], &sync[1], GRU_BLOCKS);
    }
}

// ---------------------------------------------------------------------------
// Kernel 4: final FC [B,H]x[H,2] + ES denormalization with st1 columns 1..2.
// ---------------------------------------------------------------------------
__global__ void fc_kernel(const float* __restrict__ h,
                          const float* __restrict__ Wfc,
                          const float* __restrict__ bfc,
                          const float* __restrict__ st1,
                          float* __restrict__ out) {
    int gid = blockIdx.x * blockDim.x + threadIdx.x;
    if (gid >= B_ * 2) return;
    int b = gid >> 1;
    int j = gid & 1;
    const float* hr = h + (size_t)b * H_;
    const float* wr = Wfc + (size_t)j * H_;
    float acc = 0.f;
    for (int k = 0; k < H_; ++k) acc = fmaf(hr[k], wr[k], acc);
    acc += bfc[j];
    acc = (acc - ALPHA * st1[(size_t)b * I_ + 1 + j]) / (1.f - ALPHA);
    out[gid] = acc;
}

// ---------------------------------------------------------------------------
extern "C" void kernel_launch(void* const* d_in, const int* in_sizes, int n_in,
                              void* d_out, int out_size, void* d_ws, size_t ws_size,
                              hipStream_t stream) {
    const float* x   = (const float*)d_in[0];
    const float* Wih = (const float*)d_in[1];
    const float* Whh = (const float*)d_in[2];
    const float* bih = (const float*)d_in[3];
    const float* bhh = (const float*)d_in[4];
    const float* Wfc = (const float*)d_in[5];
    const float* bfc = (const float*)d_in[6];
    float* out = (float*)d_out;

    char* ws = (char*)d_ws;
    size_t off = 0;
    unsigned short* sm = (unsigned short*)(ws + off); off += (size_t)T_ * B_ * I_ * 2;  // 64 MB
    float*          hF = (float*)(ws + off);          off += (size_t)2 * B_ * H_ * 4;   // 2 MB (ping-pong)
    unsigned short* hB = (unsigned short*)(ws + off); off += (size_t)2 * B_ * H_ * 2;   // 1 MB (ping-pong)
    float*         st1 = (float*)(ws + off);          off += (size_t)B_ * I_ * 4;       // 128 KB
    int*          sync = (int*)(ws + off);            off += 256;

    ema_kernel<<<dim3((B_ * I_) / 256), dim3(256), 0, stream>>>(x, sm, st1);
    init_kernel<<<dim3((B_ * H_ + 255) / 256), dim3(256), 0, stream>>>(hF, hB, sync);
    gru_kernel<<<dim3(GRU_BLOCKS), dim3(256), 0, stream>>>(sm, Wih, Whh, bih, bhh,
                                                           hF, hB, sync);
    fc_kernel<<<dim3(2), dim3(256), 0, stream>>>(hF, Wfc, bfc, st1, out);
}
